// BahdanauAttention_60842506715342
// MI455X (gfx1250) — matrix operand order repaired and checked by reading, not compiled, not run on hardware
//
#include <hip/hip_runtime.h>
#include <math.h>

typedef _Float16 v16h __attribute__((ext_vector_type(16)));
typedef _Float16 v8h  __attribute__((ext_vector_type(8)));
typedef float    v8f  __attribute__((ext_vector_type(8)));
typedef unsigned int v4u __attribute__((ext_vector_type(4)));
typedef int v8i __attribute__((ext_vector_type(8)));
typedef int v4i __attribute__((ext_vector_type(4)));

#define B_ 32
#define S_ 2048
#define Q_ 1024
#define I_ 1024
#define A_ 1024

// U slab staged in LDS: 16 columns (rows of Ut) x 512 K-halves, each row
// padded by 16B -> row stride 1040B = 260 dwords == 4 mod 64 banks.
#define ROWH  520                 // halves per padded LDS row
#define SLABH (16 * ROWH)         // 8320 halves = 16640 B per buffer

#if __has_builtin(__builtin_amdgcn_tensor_load_to_lds)
#define HAVE_TDM 1
#else
#define HAVE_TDM 0
#endif

// ---------------- Kernel 1: U [I,A] fp32 -> Ut [A,I] f16 (transposed) -------
__global__ void k_prep_ut(const float* __restrict__ U, _Float16* __restrict__ Ut) {
    int idx = blockIdx.x * 256 + threadIdx.x;   // A_*I_ threads
    int a = idx & (A_ - 1);                     // coalesced reads over a
    int i = idx >> 10;
    Ut[(size_t)a * I_ + i] = (_Float16)U[(size_t)i * A_ + a];
}

// ---------------- Kernel 2: pq[b,a] = sum_q queries[b,q] * W[q,a] -----------
__global__ void k_pq(const float* __restrict__ q, const float* __restrict__ W,
                     float* __restrict__ pq) {
    int b = blockIdx.x;
    int t = threadIdx.x;                        // 256 threads
    float a0 = 0.f, a1 = 0.f, a2 = 0.f, a3 = 0.f;
    const float* qb = q + (size_t)b * Q_;
    for (int k = 0; k < Q_; ++k) {
        float qv = qb[k];
        const float* Wr = W + (size_t)k * A_;
        a0 += qv * Wr[t];
        a1 += qv * Wr[t + 256];
        a2 += qv * Wr[t + 512];
        a3 += qv * Wr[t + 768];
    }
    float* o = pq + (size_t)b * A_;
    o[t] = a0; o[t + 256] = a1; o[t + 512] = a2; o[t + 768] = a3;
}

#if HAVE_TDM
// Issue one TDM load of a 16x512-half tile of Ut into LDS (padded rows).
// D# bitfields per CDNA5 ISA ch.8 (groups 0/1; groups 2/3 zero for 2D).
__device__ __forceinline__ void tdm_stage(const _Float16* gsrc, unsigned lds_off) {
    unsigned long long ga = (unsigned long long)(uintptr_t)gsrc;
    v4u g0;
    g0.x = 1u;                                          // count = 1 (valid)
    g0.y = lds_off;                                     // lds_addr (bytes)
    g0.z = (unsigned)ga;                                // global_addr[31:0]
    g0.w = (unsigned)((ga >> 32) & 0x01FFFFFFull)       // global_addr[56:32]
         | (2u << 30);                                  // type = 2 (image)
    v8i g1;
    g1[0] = (1 << 16)        // data_size = 2 bytes
          | (1 << 20)        // pad_enable
          | (7 << 22)        // pad_interval: 256 dwords = 1024 B (one row)
          | (3 << 25);       // pad_amount: 4 dwords = 16 B
    g1[1] = (int)(512u << 16);   // tensor_dim0 = 512 (lo16)
    g1[2] = (int)(16u << 16);    // tensor_dim0 hi = 0 | tensor_dim1 = 16 (lo16)
    g1[3] = (int)(512u << 16);   // tensor_dim1 hi = 0 | tile_dim0 = 512
    g1[4] = 16;                  // tile_dim1 = 16 | tile_dim2 = 0
    g1[5] = 1024;                // tensor_dim0_stride = 1024 elements (lo32)
    g1[6] = 0;                   // stride0 hi | stride1 lo
    g1[7] = 0;                   // stride1 hi
    v4i z4 = {0, 0, 0, 0};
#if __clang_major__ >= 23
    v8i z8 = {0, 0, 0, 0, 0, 0, 0, 0};
    __builtin_amdgcn_tensor_load_to_lds(g0, g1, z4, z4, z8, 0);
#else
    __builtin_amdgcn_tensor_load_to_lds(g0, g1, z4, z4, 0);
#endif
}
#else
// Fallback: cooperative copy of one 16x512-half tile into padded LDS rows.
__device__ __forceinline__ void stage_fb(_Float16* dst, const _Float16* gsrc, int tid) {
    for (int c = tid; c < 16 * 64; c += 128) {          // v8h chunks
        int r = c >> 6, h = (c & 63) * 8;
        *(v8h*)(dst + r * ROWH + h) = *(const v8h*)(gsrc + (size_t)r * I_ + h);
    }
}
#endif

// ---------------- Kernel 3: WMMA reactivity --------------------------------
// 4 waves/block, 16 item-rows per wave. A-fragments for all K=1024 live in
// 256 VGPRs per wave; U slabs double-buffered in LDS via TDM, shared by all
// waves (4x less L2 traffic than per-wave Ut streaming).
__global__ __launch_bounds__(128) void k_react(
    const float* __restrict__ items, const _Float16* __restrict__ Ut,
    const float* __restrict__ pq, const float* __restrict__ vvec,
    const int* __restrict__ wmask, float* __restrict__ react) {

    __shared__ alignas(16) _Float16 Usl[2 * SLABH];     // 33280 B

    const int tid  = threadIdx.x;
    const int wave = tid >> 5;
    const int lane = tid & 31;
    const int L16  = lane & 15;
    const int hi   = lane >> 4;

    const int row_base = blockIdx.x * 64 + wave * 16;   // row in [0, B_*S_)
    const int b  = row_base >> 11;                      // S_ = 2048
    const int s0 = row_base & (S_ - 1);

#if HAVE_TDM
    const unsigned lds_base = (unsigned)(uintptr_t)(void*)&Usl[0];
    if (wave == 0) tdm_stage(Ut, lds_base);             // prefetch slab 0
#endif

    // Build A fragments (ISA 16-bit A 16x32 layout: halves 0..7 = K hi*8..,
    // halves 8..15 = K 16+hi*8..) straight from global fp32, converted.
    v16h afrag[32];
    const float* rowp = items + (size_t)(row_base + L16) * I_;
#pragma unroll
    for (int kc = 0; kc < 32; ++kc) {
        const float* p0 = rowp + kc * 32 + hi * 8;
        const float* p1 = rowp + kc * 32 + 16 + hi * 8;
        float4 f0 = *(const float4*)(p0);
        float4 f1 = *(const float4*)(p0 + 4);
        float4 f2 = *(const float4*)(p1);
        float4 f3 = *(const float4*)(p1 + 4);
        afrag[kc] = (v16h){
            (_Float16)f0.x, (_Float16)f0.y, (_Float16)f0.z, (_Float16)f0.w,
            (_Float16)f1.x, (_Float16)f1.y, (_Float16)f1.z, (_Float16)f1.w,
            (_Float16)f2.x, (_Float16)f2.y, (_Float16)f2.z, (_Float16)f2.w,
            (_Float16)f3.x, (_Float16)f3.y, (_Float16)f3.z, (_Float16)f3.w };
    }

#if HAVE_TDM
    __builtin_amdgcn_s_wait_tensorcnt(0);
#else
    stage_fb(&Usl[0], Ut, tid);
#endif
    __syncthreads();

    float racc[8] = {0.f, 0.f, 0.f, 0.f, 0.f, 0.f, 0.f, 0.f};
    int buf = 0;
    for (int at = 0; at < A_ / 16; ++at) {
        v8f acc = {};
#pragma unroll
        for (int kh = 0; kh < 2; ++kh) {
            const int p = at * 2 + kh;                  // current phase
            if (p < 127) {                              // prefetch next slab
                const int pn = p + 1;
                const _Float16* gsrc =
                    Ut + (size_t)(pn >> 1) * (16 * I_) + (size_t)(pn & 1) * 512;
#if HAVE_TDM
                if (wave == 0)
                    tdm_stage(gsrc, lds_base + (unsigned)((buf ^ 1) * SLABH * 2));
#else
                stage_fb(&Usl[(buf ^ 1) * SLABH], gsrc, tid);
#endif
            }
            // 16 WMMA on the current slab. B-frag: lane = column L16,
            // halves 0..7 = K hi*8.., halves 8..15 = K 16+hi*8.. (two 16B LDS loads).
            const _Float16* bbase = &Usl[buf * SLABH + L16 * ROWH + hi * 8];
#pragma unroll
            for (int j = 0; j < 16; ++j) {
                v8h b0 = *(const v8h*)(bbase + j * 32);
                v8h b1 = *(const v8h*)(bbase + j * 32 + 16);
                v16h bf = __builtin_shufflevector(b0, b1,
                    0, 1, 2, 3, 4, 5, 6, 7, 8, 9, 10, 11, 12, 13, 14, 15);
                acc = __builtin_amdgcn_wmma_f32_16x16x32_f16(
                    false, afrag[kh * 16 + j], false, bf, (short)0, acc,
                    false, false);
            }
#if HAVE_TDM
            __builtin_amdgcn_s_wait_tensorcnt(0);       // next slab landed
#endif
            __syncthreads();
            buf ^= 1;
        }
        // D layout: lane -> N = L16 ; VGPR r -> M = r + 8*hi.
        const float pqv = pq[(size_t)b * A_ + at * 16 + L16];
        const float vv  = vvec[at * 16 + L16];
#pragma unroll
        for (int r = 0; r < 8; ++r)
            racc[r] += tanhf(acc[r] + pqv) * vv;
    }

    // Reduce over the 16 lanes of each half (xor masks stay within halves).
#pragma unroll
    for (int r = 0; r < 8; ++r) {
        float x = racc[r];
        x += __shfl_xor(x, 1);
        x += __shfl_xor(x, 2);
        x += __shfl_xor(x, 4);
        x += __shfl_xor(x, 8);
        racc[r] = x;
    }
    if (L16 == 0) {
#pragma unroll
        for (int r = 0; r < 8; ++r) {
            const int m = r + 8 * hi;
            const size_t off = (size_t)b * S_ + s0 + m;
            react[off] = (wmask[off] == 0) ? -INFINITY : racc[r];
        }
    }
}

// ---------------- Kernel 4: masked softmax over S per batch ----------------
__global__ void k_softmax(const float* __restrict__ react,
                          float* __restrict__ scores) {
    const int b = blockIdx.x;
    const int t = threadIdx.x;                  // 256 threads
    __shared__ float red[256];
    const float* rb = react + (size_t)b * S_;

    float m = -INFINITY;
    for (int s = t; s < S_; s += 256) m = fmaxf(m, rb[s]);
    red[t] = m; __syncthreads();
    for (int off = 128; off; off >>= 1) {
        if (t < off) red[t] = fmaxf(red[t], red[t + off]);
        __syncthreads();
    }
    const float mx = red[0];
    __syncthreads();

    float sum = 0.f;
    for (int s = t; s < S_; s += 256) sum += expf(rb[s] - mx);
    red[t] = sum; __syncthreads();
    for (int off = 128; off; off >>= 1) {
        if (t < off) red[t] += red[t + off];
        __syncthreads();
    }
    const float inv = 1.0f / red[0];

    float* sb = scores + (size_t)b * S_;
    for (int s = t; s < S_; s += 256) sb[s] = expf(rb[s] - mx) * inv;
}

// ------- Kernel 5a: blended partials over S-chunks (deterministic) ---------
__global__ void k_blend_partial(const float* __restrict__ scores,
                                const float* __restrict__ items,
                                float* __restrict__ part) {
    const int i = blockIdx.x * 256 + threadIdx.x;   // gridDim.x = I_/256
    const int b = blockIdx.y;
    const int c = blockIdx.z;                       // 8 chunks of 256 rows
    const float* it = items + (size_t)b * S_ * I_ + (size_t)c * 256 * I_ + i;
    const float* sc = scores + (size_t)b * S_ + c * 256;
    float acc = 0.f;
    for (int s = 0; s < 256; ++s)
        acc += sc[s] * it[(size_t)s * I_];
    part[((size_t)c * B_ + b) * I_ + i] = acc;
}

// ------- Kernel 5b: reduce partials ----------------------------------------
__global__ void k_blend_reduce(const float* __restrict__ part,
                               float* __restrict__ out) {
    const int o = blockIdx.x * 256 + threadIdx.x;   // B_*I_ outputs
    float s = 0.f;
#pragma unroll
    for (int c = 0; c < 8; ++c)
        s += part[(size_t)c * B_ * I_ + o];
    out[o] = s;
}

extern "C" void kernel_launch(void* const* d_in, const int* in_sizes, int n_in,
                              void* d_out, int out_size, void* d_ws, size_t ws_size,
                              hipStream_t stream) {
    const float* queries = (const float*)d_in[0];   // [B,1,Q]
    const float* items   = (const float*)d_in[1];   // [B,S,I]
    const int*   weights = (const int*)d_in[2];     // [B,S]
    const float* W       = (const float*)d_in[3];   // [Q,A]
    const float* U       = (const float*)d_in[4];   // [I,A]
    const float* v       = (const float*)d_in[5];   // [A]

    float* out = (float*)d_out;                     // blended [B,I] ++ scores [B,S]
    float* scores = out + (size_t)B_ * I_;

    char* ws = (char*)d_ws;
    size_t off = 0;
    _Float16* Ut = (_Float16*)(ws + off); off += (size_t)A_ * I_ * 2;       // 2 MB
    float* pq    = (float*)(ws + off);    off += (size_t)B_ * A_ * 4;       // 128 KB
    float* react = (float*)(ws + off);    off += (size_t)B_ * S_ * 4;       // 256 KB
    float* part  = (float*)(ws + off);    off += (size_t)8 * B_ * I_ * 4;   // 1 MB

    k_prep_ut<<<(A_ * I_) / 256, 256, 0, stream>>>(U, Ut);
    k_pq<<<B_, 256, 0, stream>>>(queries, W, pq);
    k_react<<<(B_ * S_) / 64, 128, 0, stream>>>(items, Ut, pq, v, weights, react);
    k_softmax<<<B_, 256, 0, stream>>>(react, scores);
    k_blend_partial<<<dim3(I_ / 256, B_, 8), 256, 0, stream>>>(scores, items, part);
    k_blend_reduce<<<(B_ * I_) / 256, 256, 0, stream>>>(part, out);
}
